// PolarMask_Head_67250597921273
// MI455X (gfx1250) — matrix operand order repaired and checked by reading, not compiled, not run on hardware
//
#include <hip/hip_runtime.h>

typedef __attribute__((ext_vector_type(16))) _Float16 v16h;
typedef __attribute__((ext_vector_type(8)))  float    v8f;

#define FEATC 256
#define KTOT  2304      // 9 * 256
#define KCHUNKS 72      // KTOT / 32

union AFrag { v16h v; unsigned u[8]; };

// ---------------------------------------------------------------------------
// Load one 16x32 A fragment (16 pixels x 32 input channels at tap (dy,dx)).
// 16-bit A layout: VGPR i holds K = 2i + (i>=4?8:0) + 8*halfwave.
// ---------------------------------------------------------------------------
__device__ __forceinline__ void load_afrag(
    AFrag& a, const _Float16* __restrict__ xin,
    int y, int x, int dy, int dx, int H, int W, int Wsh, int icb, int hs)
{
  const int ya = y + dy, xa = x + dx;
  const bool inb = (ya >= 0) && (ya < H) && (xa >= 0) && (xa < W);
  const unsigned* __restrict__ ap =
      (const unsigned*)(xin + (((ya << Wsh) + xa) << 8) + icb);
  #pragma unroll
  for (int i = 0; i < 8; ++i) {
    const int koff = 2 * i + ((i >= 4) ? 8 : 0) + (hs << 3);
    unsigned vv = 0u;
    if (inb) vv = ap[koff >> 1];
    a.u[i] = vv;
  }
}

// ---------------------------------------------------------------------------
// Tower conv, 2x2 register-blocked: one wave computes 32 pixels x 32 output
// channels = 4 accumulators, 4 v_wmma per K-chunk, 8 dword loads per WMMA.
// Writes pre-norm f16 NHWC and accumulates GroupNorm sum/sumsq.
// ---------------------------------------------------------------------------
__global__ void tower_conv_wmma2(
    const _Float16* __restrict__ xin, const _Float16* __restrict__ wB,
    _Float16* __restrict__ xout, float* __restrict__ stats,
    int H, int W, int Wsh, int Mtiles32)
{
  const int wave = blockIdx.x * (blockDim.x >> 5) + (threadIdx.x >> 5);
  if (wave >= Mtiles32 * 8) return;     // wave-uniform exit
  const int mt = wave >> 3;             // 32-pixel row block
  const int nt = wave & 7;              // 32-channel col block
  const int lane = threadIdx.x & 31;
  const int hs = lane >> 4, ml = lane & 15;

  const int pm = mt << 5;               // base pixel
  const int p0 = pm + ml, p1 = p0 + 16;
  const int y0 = p0 >> Wsh, x0 = p0 & (W - 1);
  const int y1 = p1 >> Wsh, x1 = p1 & (W - 1);
  const int nc0 = (nt << 5) + ml, nc1 = nc0 + 16;

  const unsigned* __restrict__ bp = (const unsigned*)wB;
  v8f a00 = {}, a01 = {}, a10 = {}, a11 = {};

  for (int kc = 0; kc < KCHUNKS; ++kc) {
    const int r   = kc >> 3;
    const int icb = (kc & 7) << 5;
    const int dy  = r / 3 - 1;
    const int dx  = r % 3 - 1;
    AFrag fa0, fa1, fb0, fb1;
    load_afrag(fa0, xin, y0, x0, dy, dx, H, W, Wsh, icb, hs);
    load_afrag(fa1, xin, y1, x1, dy, dx, H, W, Wsh, icb, hs);
    const int kbase = kc << 5;
    #pragma unroll
    for (int i = 0; i < 8; ++i) {       // B layout: VGPR i holds K=2i+(i>=4?8:0)+8*hs
      const int row = (kbase + 2 * i + ((i >= 4) ? 8 : 0) + (hs << 3)) >> 1;
      fb0.u[i] = bp[row * FEATC + nc0];
      fb1.u[i] = bp[row * FEATC + nc1];
    }
    if (kc + 1 < KCHUNKS)
      __builtin_prefetch(bp + (((kbase + 32) >> 1) * FEATC) + nc0, 0, 1);
    a00 = __builtin_amdgcn_wmma_f32_16x16x32_f16(false, fa0.v, false, fb0.v,
                                                 (short)0, a00, false, false);
    a01 = __builtin_amdgcn_wmma_f32_16x16x32_f16(false, fa0.v, false, fb1.v,
                                                 (short)0, a01, false, false);
    a10 = __builtin_amdgcn_wmma_f32_16x16x32_f16(false, fa1.v, false, fb0.v,
                                                 (short)0, a10, false, false);
    a11 = __builtin_amdgcn_wmma_f32_16x16x32_f16(false, fa1.v, false, fb1.v,
                                                 (short)0, a11, false, false);
  }

  float s0 = 0.f, q0 = 0.f, s1 = 0.f, q1 = 0.f;
  #pragma unroll
  for (int r = 0; r < 8; ++r) {         // C/D layout: row = r + 8*hs
    const int pa = pm + r + (hs << 3);
    const int pb = pa + 16;
    const float v00 = a00[r], v01 = a01[r], v10 = a10[r], v11 = a11[r];
    xout[(pa << 8) + nc0] = (_Float16)v00;
    xout[(pa << 8) + nc1] = (_Float16)v01;
    xout[(pb << 8) + nc0] = (_Float16)v10;
    xout[(pb << 8) + nc1] = (_Float16)v11;
    s0 += v00 + v10; q0 += v00 * v00 + v10 * v10;
    s1 += v01 + v11; q1 += v01 * v01 + v11 * v11;
  }
  // lanes {l^1,l^2,l^4} share the same GN group (oc>>3): reduce 8x
  #pragma unroll
  for (int off = 1; off < 8; off <<= 1) {
    s0 += __shfl_xor(s0, off, 32);  q0 += __shfl_xor(q0, off, 32);
    s1 += __shfl_xor(s1, off, 32);  q1 += __shfl_xor(q1, off, 32);
  }
  if ((lane & 7) == 0) {
    atomicAdd(&stats[(nc0 >> 3) * 2 + 0], s0);
    atomicAdd(&stats[(nc0 >> 3) * 2 + 1], q0);
    atomicAdd(&stats[(nc1 >> 3) * 2 + 0], s1);
    atomicAdd(&stats[(nc1 >> 3) * 2 + 1], q1);
  }
}

// ---------------------------------------------------------------------------
// 1x1-tile conv for prediction heads (small N): bias + optional exp(x*scale),
// f32 NCHW output into d_out.
// ---------------------------------------------------------------------------
__global__ void head_conv_wmma(
    const _Float16* __restrict__ xin, const _Float16* __restrict__ wB,
    const float* __restrict__ bias, float* __restrict__ out,
    const float* __restrict__ scale_ptr, int H, int W, int Wsh,
    int Npad, int Cout, int do_exp, int Mtiles)
{
  const int NT = Npad >> 4;
  const int wave = blockIdx.x * (blockDim.x >> 5) + (threadIdx.x >> 5);
  if (wave >= Mtiles * NT) return;
  const int mt = wave / NT, nt = wave % NT;
  const int lane = threadIdx.x & 31;
  const int hs = lane >> 4, ml = lane & 15;
  const int p0 = (mt << 4) + ml;
  const int y0 = p0 >> Wsh, x0 = p0 & (W - 1);
  const int ncol = (nt << 4) + ml;
  const unsigned* __restrict__ bp = (const unsigned*)wB;
  v8f acc = {};
  for (int kc = 0; kc < KCHUNKS; ++kc) {
    const int r   = kc >> 3;
    const int icb = (kc & 7) << 5;
    const int dy  = r / 3 - 1;
    const int dx  = r % 3 - 1;
    AFrag fa, fb;
    load_afrag(fa, xin, y0, x0, dy, dx, H, W, Wsh, icb, hs);
    const int kbase = kc << 5;
    #pragma unroll
    for (int i = 0; i < 8; ++i) {
      const int row = (kbase + 2 * i + ((i >= 4) ? 8 : 0) + (hs << 3)) >> 1;
      fb.u[i] = bp[row * Npad + ncol];
    }
    acc = __builtin_amdgcn_wmma_f32_16x16x32_f16(false, fa.v, false, fb.v,
                                                 (short)0, acc, false, false);
  }
  if (ncol >= Cout) return;
  const int HW = H * W;
  const float bsv = bias[ncol];
  const float sc = do_exp ? scale_ptr[0] : 0.f;
  #pragma unroll
  for (int r = 0; r < 8; ++r) {
    const int p = (mt << 4) + r + (hs << 3);
    float v = acc[r] + bsv;
    if (do_exp) v = expf(v * sc);
    out[(size_t)ncol * HW + p] = v;
  }
}

// GroupNorm (from accumulated stats) + ReLU, f16 NHWC -> f16 NHWC.
__global__ void gn_relu_f16(
    const _Float16* __restrict__ xin, _Float16* __restrict__ xout,
    const float* __restrict__ stats, const float* __restrict__ gamma,
    const float* __restrict__ beta, int HW)
{
  const int e = blockIdx.x * blockDim.x + threadIdx.x;
  if (e >= HW * FEATC) return;
  const int c = e & 255;
  const int g = c >> 3;
  const float cnt = 8.0f * (float)HW;
  const float m = stats[g * 2 + 0] / cnt;
  const float var = stats[g * 2 + 1] / cnt - m * m;
  const float rs = rsqrtf(var + 1e-5f);
  float v = ((float)xin[e] - m) * rs * gamma[c] + beta[c];
  v = v > 0.f ? v : 0.f;
  xout[e] = (_Float16)v;
}

// f32 NCHW input -> f16 NHWC.
__global__ void nchw_to_nhwc_f16(const float* __restrict__ fin,
                                 _Float16* __restrict__ fout, int HW)
{
  const int e = blockIdx.x * blockDim.x + threadIdx.x;
  if (e >= HW * FEATC) return;
  const int c = e & 255;
  const int p = e >> 8;
  fout[e] = (_Float16)fin[(size_t)c * HW + p];
}

// Tower weights [4][256][256][3][3] f32 -> per-layer packed B f16 [k/2][256][2],
// with k = (ky*3+kx)*256 + ic.
__global__ void pack_tower_w(const float* __restrict__ src,
                             _Float16* __restrict__ dst, int total)
{
  const int idx = blockIdx.x * blockDim.x + threadIdx.x;
  if (idx >= total) return;
  const int l  = idx / (FEATC * KTOT);
  const int rm = idx % (FEATC * KTOT);
  const int oc = rm / KTOT;
  const int k  = rm % KTOT;
  const int r  = k >> 8;
  const int ic = k & 255;
  const float w = src[((size_t)((l * FEATC + oc) * FEATC + ic)) * 9 + r];
  dst[(size_t)l * (KTOT * FEATC) + (size_t)(k >> 1) * 512 + oc * 2 + (k & 1)] =
      (_Float16)w;
}

// Head weights [Cout][256][3][3] f32 -> packed B f16 [k/2][Npad][2], zero-pad N.
__global__ void pack_head_w(const float* __restrict__ src,
                            _Float16* __restrict__ dst,
                            int Cout, int Npad, int total)
{
  const int idx = blockIdx.x * blockDim.x + threadIdx.x;
  if (idx >= total) return;
  const int n = idx / KTOT;
  const int k = idx % KTOT;
  const int r = k >> 8, ic = k & 255;
  float w = 0.f;
  if (n < Cout) w = src[((size_t)(n * FEATC + ic)) * 9 + r];
  dst[(size_t)(k >> 1) * (Npad * 2) + n * 2 + (k & 1)] = (_Float16)w;
}

// ---------------------------------------------------------------------------
extern "C" void kernel_launch(void* const* d_in, const int* in_sizes, int n_in,
                              void* d_out, int out_size, void* d_ws, size_t ws_size,
                              hipStream_t stream)
{
  (void)in_sizes; (void)n_in; (void)out_size; (void)ws_size;
  const float* feats[5];
  for (int l = 0; l < 5; ++l) feats[l] = (const float*)d_in[l];
  const float* tw[3]  = {(const float*)d_in[5],  (const float*)d_in[8],  (const float*)d_in[11]};
  const float* tg[3]  = {(const float*)d_in[6],  (const float*)d_in[9],  (const float*)d_in[12]};
  const float* tbt[3] = {(const float*)d_in[7],  (const float*)d_in[10], (const float*)d_in[13]};
  const float* hw_[4] = {(const float*)d_in[14], (const float*)d_in[16],
                         (const float*)d_in[18], (const float*)d_in[20]}; // pcls,preg,pmsk,pctr
  const float* hb_[4] = {(const float*)d_in[15], (const float*)d_in[17],
                         (const float*)d_in[19], (const float*)d_in[21]};
  const float* sc_bbox = (const float*)d_in[22];
  const float* sc_mask = (const float*)d_in[23];

  // Workspace layout (bytes): stats | tower W f16 | head W f16 | actIn | bufA | bufB
  char* ws = (char*)d_ws;
  float*    stats = (float*)ws;                         // 60 slots * 64 floats
  _Float16* wT    = (_Float16*)(ws + 16384);            // 3*4*2304*256 halves
  _Float16* wH    = (_Float16*)(ws + 14172160);         // 2304*160 halves
  _Float16* actIn = (_Float16*)(ws + 14909440);         // 256*40960 halves
  _Float16* bufA  = (_Float16*)(ws + 35880960);
  _Float16* bufB  = (_Float16*)(ws + 56852480);

  hipMemsetAsync(stats, 0, 60 * 64 * sizeof(float), stream);

  // Pack weights into WMMA B-matrix layouts.
  for (int t = 0; t < 3; ++t) {
    const int tot = 4 * FEATC * KTOT;
    pack_tower_w<<<(tot + 255) / 256, 256, 0, stream>>>(
        tw[t], wT + (size_t)t * 4 * KTOT * FEATC, tot);
  }
  const int    hNpad[4] = {80, 16, 48, 16};
  const int    hCout[4] = {80, 4, 36, 1};
  const size_t hOff[4]  = {0, (size_t)KTOT * 80, (size_t)KTOT * 96, (size_t)KTOT * 144};
  for (int h = 0; h < 4; ++h) {
    const int tot = hNpad[h] * KTOT;
    pack_head_w<<<(tot + 255) / 256, 256, 0, stream>>>(
        hw_[h], wH + hOff[h], hCout[h], hNpad[h], tot);
  }

  const int Hs[5] = {160, 80, 40, 20, 10};
  const int Wd[5] = {256, 128, 64, 32, 16};
  const int prefix[5] = {0, 40960, 51200, 53760, 54400};
  float* out = (float*)d_out;

  for (int l = 0; l < 5; ++l) {
    const int H = Hs[l], W = Wd[l], HW = H * W, Wsh = 8 - l;
    const int Mtiles = HW / 16, Mtiles32 = HW / 32;
    const int eblocks = (HW * FEATC + 255) / 256;
    nchw_to_nhwc_f16<<<eblocks, 256, 0, stream>>>(feats[l], actIn, HW);

    for (int t = 0; t < 3; ++t) {
      const _Float16* src = actIn;
      for (int s = 0; s < 4; ++s) {
        const int slot = (l * 3 + t) * 4 + s;
        const int tiles = Mtiles32 * 8;
        tower_conv_wmma2<<<(tiles + 7) / 8, 256, 0, stream>>>(
            src, wT + (size_t)(t * 4 + s) * KTOT * FEATC, bufA,
            stats + slot * 64, H, W, Wsh, Mtiles32);
        gn_relu_f16<<<eblocks, 256, 0, stream>>>(
            bufA, bufB, stats + slot * 64, tg[t] + s * 256, tbt[t] + s * 256, HW);
        src = bufB;
      }
      if (t == 0) {       // cls tower -> pcls + pctr
        int tiles = Mtiles * (hNpad[0] >> 4);
        head_conv_wmma<<<(tiles + 7) / 8, 256, 0, stream>>>(
            bufB, wH + hOff[0], hb_[0], out + (size_t)80 * prefix[l],
            nullptr, H, W, Wsh, hNpad[0], hCout[0], 0, Mtiles);
        tiles = Mtiles * (hNpad[3] >> 4);
        head_conv_wmma<<<(tiles + 7) / 8, 256, 0, stream>>>(
            bufB, wH + hOff[3], hb_[3], out + 4583040 + (size_t)prefix[l],
            nullptr, H, W, Wsh, hNpad[3], hCout[3], 0, Mtiles);
      } else if (t == 1) { // reg tower -> preg (exp * scales_bbox[l])
        const int tiles = Mtiles * (hNpad[1] >> 4);
        head_conv_wmma<<<(tiles + 7) / 8, 256, 0, stream>>>(
            bufB, wH + hOff[1], hb_[1], out + 4364800 + (size_t)4 * prefix[l],
            sc_bbox + l, H, W, Wsh, hNpad[1], hCout[1], 1, Mtiles);
      } else {             // msk tower -> pmsk (exp * scales_mask[l])
        const int tiles = Mtiles * (hNpad[2] >> 4);
        head_conv_wmma<<<(tiles + 7) / 8, 256, 0, stream>>>(
            bufB, wH + hOff[2], hb_[2], out + 4637600 + (size_t)36 * prefix[l],
            sc_mask + l, H, W, Wsh, hNpad[2], hCout[2], 1, Mtiles);
      }
    }
  }
}
